// GATv2_38104949850572
// MI455X (gfx1250) — compile-verified, matrix operand order checked by reading, not run on hardware
//
#include <hip/hip_runtime.h>
#include <math.h>

// ---------------------------------------------------------------------------
// GATv2 (2 layers, H=4/C=64 then H=1/C=64) + mean pool + linear, for gfx1250.
// GEMMs: v_wmma_f32_16x16x32_bf16 with bf16 hi/lo split (~f32 accuracy, f32
// accumulate). A/B are pre-packed into WMMA fragment order so the hot loop is
// pure b128 loads + WMMA. Edge phase: wave-per-edge, native f32 atomics.
// ---------------------------------------------------------------------------

typedef __bf16 bf16_t;
typedef __attribute__((ext_vector_type(16))) __bf16        v16bf;
typedef __attribute__((ext_vector_type(8)))  float         v8f;
typedef __attribute__((ext_vector_type(4)))  unsigned int  v4u;

#define KN      50000
#define KE      800000
#define KETOT   (KE + KN)
#define KFIN    128
#define KHC     256   // H*C layer 1
#define KC2     64    // C layer 2
#define KG      8

static __device__ __forceinline__ unsigned int packbf2(float x0, float x1,
                                                       unsigned int* lo_out) {
  bf16_t h0 = (bf16_t)x0, h1 = (bf16_t)x1;
  bf16_t l0 = (bf16_t)(x0 - (float)h0), l1 = (bf16_t)(x1 - (float)h1);
  unsigned short uh0 = __builtin_bit_cast(unsigned short, h0);
  unsigned short uh1 = __builtin_bit_cast(unsigned short, h1);
  unsigned short ul0 = __builtin_bit_cast(unsigned short, l0);
  unsigned short ul1 = __builtin_bit_cast(unsigned short, l1);
  *lo_out = (unsigned int)ul0 | ((unsigned int)ul1 << 16);
  return (unsigned int)uh0 | ((unsigned int)uh1 << 16);
}

// ---------------------------------------------------------------- fill
__global__ void fillk(float* __restrict__ p, float v, int n) {
  int i = blockIdx.x * blockDim.x + threadIdx.x;
  if (i < n) p[i] = v;
}

// -------------------------------------------------- fragment pack kernels
// A fragment order (ISA 7.12.2, 16-bit A 16x32): tile = mt*(K/32)+kt; within a
// tile: lane owns row (lane&15), k-base 8*(lane>>4); dword j packs K pair.
// Storage: dword index = (tile*32 + lane)*8 + j  -> lane stride 32B, b128-able.
static __device__ __forceinline__ int frag_k_of_j(int j) {
  return (j < 4) ? 2 * j : 16 + 2 * (j - 4);
}

__global__ void pack_a_f32(const float* __restrict__ in, unsigned int* __restrict__ hi,
                           unsigned int* __restrict__ lo, int M, int K) {
  int i = blockIdx.x * blockDim.x + threadIdx.x;          // one dword pair
  int D = (M * K) >> 1;
  if (i >= D) return;
  int tile = i >> 8, within = i & 255;
  int lane = within >> 3, j = within & 7;
  int ktiles = K >> 5;
  int mt = tile / ktiles, kt = tile % ktiles;
  int m = (mt << 4) + (lane & 15);
  int k = (kt << 5) + ((lane >> 4) << 3) + frag_k_of_j(j);
  size_t s = (size_t)m * K + k;
  unsigned int l;
  unsigned int h = packbf2(in[s], in[s + 1], &l);
  hi[i] = h; lo[i] = l;
}

// B fragment: tile = kt*(N/16)+nt; lane owns column (lane&15), same k packing.
__global__ void pack_b_f32(const float* __restrict__ in, unsigned int* __restrict__ hi,
                           unsigned int* __restrict__ lo, int K, int Nn) {
  int i = blockIdx.x * blockDim.x + threadIdx.x;
  int D = (K * Nn) >> 1;
  if (i >= D) return;
  int tile = i >> 8, within = i & 255;
  int lane = within >> 3, j = within & 7;
  int ntiles = Nn >> 4;
  int kt = tile / ntiles, nt = tile % ntiles;
  int n = (nt << 4) + (lane & 15);
  int k = (kt << 5) + ((lane >> 4) << 3) + frag_k_of_j(j);
  unsigned int l;
  unsigned int h = packbf2(in[(size_t)k * Nn + n], in[(size_t)(k + 1) * Nn + n], &l);
  hi[i] = h; lo[i] = l;
}

// bias + relu + pack straight into A-fragment order (feeds layer-2 GEMM)
__global__ void bias_relu_pack_a(const float* __restrict__ in, const float* __restrict__ b,
                                 unsigned int* __restrict__ hi, unsigned int* __restrict__ lo,
                                 int M, int K) {
  int i = blockIdx.x * blockDim.x + threadIdx.x;
  int D = (M * K) >> 1;
  if (i >= D) return;
  int tile = i >> 8, within = i & 255;
  int lane = within >> 3, j = within & 7;
  int ktiles = K >> 5;
  int mt = tile / ktiles, kt = tile % ktiles;
  int m = (mt << 4) + (lane & 15);
  int k = (kt << 5) + ((lane >> 4) << 3) + frag_k_of_j(j);
  size_t s = (size_t)m * K + k;
  float x0 = in[s] + b[k],     x1 = in[s + 1] + b[k + 1];
  x0 = x0 > 0.f ? x0 : 0.f;    x1 = x1 > 0.f ? x1 : 0.f;
  unsigned int l;
  unsigned int h = packbf2(x0, x1, &l);
  hi[i] = h; lo[i] = l;
}

// --------------------------------------------------------------- WMMA GEMM
// C[M,N] = A[M,K]*B[K,N] from pre-packed fragments. One wave per 16(M)x64(N)
// strip: 4 f32 accumulators, A fragment reused 4x, 12 WMMAs per k-step.
__global__ void gemm_wmma_frag(const v4u* __restrict__ Ah, const v4u* __restrict__ Al,
                               const v4u* __restrict__ Bh, const v4u* __restrict__ Bl,
                               float* __restrict__ C, int M, int Nn, int K) {
  int wave = (blockIdx.x * blockDim.x + threadIdx.x) >> 5;
  int lane = threadIdx.x & 31;
  int ngroups = Nn >> 6;                           // groups of 4 N-tiles
  int total = (M >> 4) * ngroups;
  if (wave >= total) return;                       // wave-uniform: EXEC all-1s
  int mt  = wave / ngroups;
  int nt0 = (wave % ngroups) << 2;
  int ktiles = K >> 5, ntiles = Nn >> 4;

  union F { v16bf v; v4u q[2]; };
  F ah, al, bh, bl;
  v8f c0 = {}, c1 = {}, c2 = {}, c3 = {};

  for (int kt = 0; kt < ktiles; ++kt) {
    size_t abase = ((size_t)(mt * ktiles + kt) * 32 + lane) * 2;
    ah.q[0] = Ah[abase]; ah.q[1] = Ah[abase + 1];
    al.q[0] = Al[abase]; al.q[1] = Al[abase + 1];
#pragma unroll
    for (int w = 0; w < 4; ++w) {
      size_t bbase = ((size_t)(kt * ntiles + nt0 + w) * 32 + lane) * 2;
      bh.q[0] = Bh[bbase]; bh.q[1] = Bh[bbase + 1];
      bl.q[0] = Bl[bbase]; bl.q[1] = Bl[bbase + 1];
      v8f acc = (w == 0) ? c0 : (w == 1) ? c1 : (w == 2) ? c2 : c3;
      acc = __builtin_amdgcn_wmma_f32_16x16x32_bf16(false, ah.v, false, bh.v, (short)0, acc, false, false);
      acc = __builtin_amdgcn_wmma_f32_16x16x32_bf16(false, ah.v, false, bl.v, (short)0, acc, false, false);
      acc = __builtin_amdgcn_wmma_f32_16x16x32_bf16(false, al.v, false, bh.v, (short)0, acc, false, false);
      if (w == 0) c0 = acc; else if (w == 1) c1 = acc; else if (w == 2) c2 = acc; else c3 = acc;
    }
  }

  int half = lane >> 4, m0 = mt << 4;
  union { v8f v; float f[8]; } o;
#pragma unroll
  for (int w = 0; w < 4; ++w) {
    o.v = (w == 0) ? c0 : (w == 1) ? c1 : (w == 2) ? c2 : c3;
    int n = ((nt0 + w) << 4) + (lane & 15);
#pragma unroll
    for (int r = 0; r < 8; ++r)                    // C layout: VGPR r -> row r+8*half
      C[(size_t)(m0 + r + 8 * half) * Nn + n] = o.f[r];
  }
}

// ------------------------------------------------------------ edge kernels
template <int FEATS, int HEADS>
__global__ void edge_score(const float* __restrict__ xl, const float* __restrict__ xr,
                           const float* __restrict__ att, const int* __restrict__ ei,
                           float* __restrict__ score, float* __restrict__ smax) {
  const int PER = FEATS / 32;
  const int CH  = FEATS / HEADS;
  int e    = (blockIdx.x * blockDim.x + threadIdx.x) >> 5;
  int lane = threadIdx.x & 31;
  if (e >= KETOT) return;
  int src, dst;
  if (e < KE) { src = ei[e]; dst = ei[KE + e]; } else { src = dst = e - KE; }
  const float* pl = xl + (size_t)src * FEATS;
  const float* pr = xr + (size_t)dst * FEATS;
  float hacc[HEADS];
#pragma unroll
  for (int h = 0; h < HEADS; ++h) hacc[h] = 0.f;
#pragma unroll
  for (int t = 0; t < PER; ++t) {
    int f = lane + 32 * t;
    float v = pl[f] + pr[f];
    v = v > 0.f ? v : 0.2f * v;
    hacc[(t * 32) / CH] += v * att[f];             // head index is unroll-constant
  }
#pragma unroll
  for (int h = 0; h < HEADS; ++h) {
    float s = hacc[h];
#pragma unroll
    for (int off = 16; off; off >>= 1) s += __shfl_xor(s, off, 32);
    if (lane == 0) {
      score[(size_t)e * HEADS + h] = s;
      __hip_atomic_fetch_max(&smax[(size_t)dst * HEADS + h], s,
                             __ATOMIC_RELAXED, __HIP_MEMORY_SCOPE_AGENT);
    }
  }
}

template <int HEADS>
__global__ void edge_exp(float* __restrict__ score, const float* __restrict__ smax,
                         float* __restrict__ denom, const int* __restrict__ ei) {
  int e = blockIdx.x * blockDim.x + threadIdx.x;
  if (e >= KETOT) return;
  int dst = (e < KE) ? ei[KE + e] : e - KE;
#pragma unroll
  for (int h = 0; h < HEADS; ++h) {
    float ex = __expf(score[(size_t)e * HEADS + h] - smax[(size_t)dst * HEADS + h]);
    score[(size_t)e * HEADS + h] = ex;
    atomicAdd(&denom[(size_t)dst * HEADS + h], ex);
  }
}

template <int FEATS, int HEADS>
__global__ void edge_aggr(const float* __restrict__ ex, const float* __restrict__ denom,
                          const float* __restrict__ xl, const int* __restrict__ ei,
                          float* __restrict__ out) {
  const int PER = FEATS / 32;
  const int CH  = FEATS / HEADS;
  int e    = (blockIdx.x * blockDim.x + threadIdx.x) >> 5;
  int lane = threadIdx.x & 31;
  if (e >= KETOT) return;
  int src, dst;
  if (e < KE) { src = ei[e]; dst = ei[KE + e]; } else { src = dst = e - KE; }
  float a = 0.f;
  if (lane < HEADS)
    a = ex[(size_t)e * HEADS + lane] / (denom[(size_t)dst * HEADS + lane] + 1e-16f);
  float al[HEADS];
#pragma unroll
  for (int h = 0; h < HEADS; ++h) al[h] = __shfl(a, h, 32);
  const float* pl = xl + (size_t)src * FEATS;
  float*       po = out + (size_t)dst * FEATS;
#pragma unroll
  for (int t = 0; t < PER; ++t) {
    int f = lane + 32 * t;
    atomicAdd(&po[f], pl[f] * al[(t * 32) / CH]);
  }
}

// -------------------------------------------------------------- pool + head
__global__ void pool_accum(const float* __restrict__ h, const float* __restrict__ b2,
                           const int* __restrict__ batch, float* __restrict__ pooled,
                           float* __restrict__ cnt) {
  int i = blockIdx.x * blockDim.x + threadIdx.x;
  int n = i >> 6, c = i & 63;
  if (n >= KN) return;
  int g = batch[n];
  atomicAdd(&pooled[g * 64 + c], h[(size_t)n * 64 + c] + b2[c]);
  if (c == 0) atomicAdd(&cnt[g], 1.0f);
}

__global__ void final_head(const float* __restrict__ pooled, const float* __restrict__ cnt,
                           const float* __restrict__ Wo, const float* __restrict__ bo,
                           float* __restrict__ out) {
  int g = threadIdx.x >> 5, lane = threadIdx.x & 31;
  if (g >= KG) return;
  float cg = cnt[g]; cg = cg > 1.f ? cg : 1.f;
  float s = pooled[g * 64 + lane]      * Wo[lane]
          + pooled[g * 64 + lane + 32] * Wo[lane + 32];
#pragma unroll
  for (int off = 16; off; off >>= 1) s += __shfl_xor(s, off, 32);
  if (lane == 0) out[g] = s / cg + bo[0];
}

// -------------------------------------------------------------------- host
static inline int ceil_div(long long a, long long b) { return (int)((a + b - 1) / b); }

extern "C" void kernel_launch(void* const* d_in, const int* in_sizes, int n_in,
                              void* d_out, int out_size, void* d_ws, size_t ws_size,
                              hipStream_t stream) {
  (void)in_sizes; (void)n_in; (void)out_size; (void)ws_size;
  const float* x    = (const float*)d_in[0];
  const int*   ei   = (const int*)d_in[1];
  const int*   batc = (const int*)d_in[2];
  const float* Wl1  = (const float*)d_in[3];
  const float* Wr1  = (const float*)d_in[4];
  const float* att1 = (const float*)d_in[5];
  const float* b1   = (const float*)d_in[6];
  const float* Wl2  = (const float*)d_in[7];
  const float* Wr2  = (const float*)d_in[8];
  const float* att2 = (const float*)d_in[9];
  const float* b2   = (const float*)d_in[10];
  const float* Wo   = (const float*)d_in[11];
  const float* bo   = (const float*)d_in[12];
  float* out = (float*)d_out;

  // workspace bump allocator (256B aligned)
  char* wp = (char*)d_ws;
  auto alloc = [&](size_t bytes) -> void* {
    void* p = (void*)wp;
    wp += (bytes + 255) & ~(size_t)255;
    return p;
  };
  // fragment buffers hold (dim0*dim1/2) dwords each
  unsigned int* a1h = (unsigned int*)alloc((size_t)KN * KFIN * 2);
  unsigned int* a1l = (unsigned int*)alloc((size_t)KN * KFIN * 2);
  unsigned int* wl1h = (unsigned int*)alloc((size_t)KFIN * KHC * 2);
  unsigned int* wl1l = (unsigned int*)alloc((size_t)KFIN * KHC * 2);
  unsigned int* wr1h = (unsigned int*)alloc((size_t)KFIN * KHC * 2);
  unsigned int* wr1l = (unsigned int*)alloc((size_t)KFIN * KHC * 2);
  unsigned int* wl2h = (unsigned int*)alloc((size_t)KHC * KC2 * 2);
  unsigned int* wl2l = (unsigned int*)alloc((size_t)KHC * KC2 * 2);
  unsigned int* wr2h = (unsigned int*)alloc((size_t)KHC * KC2 * 2);
  unsigned int* wr2l = (unsigned int*)alloc((size_t)KHC * KC2 * 2);
  float*  xl1   = (float*)alloc((size_t)KN * KHC * 4);
  float*  xr1   = (float*)alloc((size_t)KN * KHC * 4);
  float*  sc1   = (float*)alloc((size_t)KETOT * 4 * 4);
  float*  smax1 = (float*)alloc((size_t)KN * 4 * 4);
  float*  den1  = (float*)alloc((size_t)KN * 4 * 4);
  float*  out1  = (float*)alloc((size_t)KN * KHC * 4);
  unsigned int* h1h = (unsigned int*)alloc((size_t)KN * KHC * 2);
  unsigned int* h1l = (unsigned int*)alloc((size_t)KN * KHC * 2);
  float*  xl2   = (float*)alloc((size_t)KN * KC2 * 4);
  float*  xr2   = (float*)alloc((size_t)KN * KC2 * 4);
  float*  sc2   = (float*)alloc((size_t)KETOT * 4);
  float*  smax2 = (float*)alloc((size_t)KN * 4);
  float*  den2  = (float*)alloc((size_t)KN * 4);
  float*  out2  = (float*)alloc((size_t)KN * KC2 * 4);
  float*  pooled= (float*)alloc((size_t)KG * KC2 * 4);
  float*  cnt   = (float*)alloc((size_t)KG * 4);

  const int T = 256;
  // --- pack inputs/weights into WMMA fragments (fused bf16 hi/lo split) ---
  pack_a_f32<<<ceil_div((long long)KN * KFIN / 2, T), T, 0, stream>>>(x, a1h, a1l, KN, KFIN);
  pack_b_f32<<<ceil_div(KFIN * KHC / 2, T), T, 0, stream>>>(Wl1, wl1h, wl1l, KFIN, KHC);
  pack_b_f32<<<ceil_div(KFIN * KHC / 2, T), T, 0, stream>>>(Wr1, wr1h, wr1l, KFIN, KHC);
  pack_b_f32<<<ceil_div(KHC * KC2 / 2, T), T, 0, stream>>>(Wl2, wl2h, wl2l, KHC, KC2);
  pack_b_f32<<<ceil_div(KHC * KC2 / 2, T), T, 0, stream>>>(Wr2, wr2h, wr2l, KHC, KC2);

  // --- layer 1 GEMMs on WMMA ---
  {
    int waves = (KN / 16) * (KHC / 64);
    int blocks = ceil_div(waves, 8);               // 8 waves / block
    gemm_wmma_frag<<<blocks, T, 0, stream>>>((const v4u*)a1h, (const v4u*)a1l,
                                             (const v4u*)wl1h, (const v4u*)wl1l,
                                             xl1, KN, KHC, KFIN);
    gemm_wmma_frag<<<blocks, T, 0, stream>>>((const v4u*)a1h, (const v4u*)a1l,
                                             (const v4u*)wr1h, (const v4u*)wr1l,
                                             xr1, KN, KHC, KFIN);
  }

  // --- layer 1 edge phase ---
  fillk<<<ceil_div(KN * 4, T), T, 0, stream>>>(smax1, -INFINITY, KN * 4);
  fillk<<<ceil_div(KN * 4, T), T, 0, stream>>>(den1, 0.f, KN * 4);
  fillk<<<ceil_div((long long)KN * KHC, T), T, 0, stream>>>(out1, 0.f, KN * KHC);
  {
    int eb = ceil_div((long long)KETOT * 32, T);
    edge_score<KHC, 4><<<eb, T, 0, stream>>>(xl1, xr1, att1, ei, sc1, smax1);
    edge_exp<4><<<ceil_div(KETOT, T), T, 0, stream>>>(sc1, smax1, den1, ei);
    edge_aggr<KHC, 4><<<eb, T, 0, stream>>>(sc1, den1, xl1, ei, out1);
  }
  bias_relu_pack_a<<<ceil_div((long long)KN * KHC / 2, T), T, 0, stream>>>(
      out1, b1, h1h, h1l, KN, KHC);

  // --- layer 2 GEMMs on WMMA ---
  {
    int waves = (KN / 16) * (KC2 / 64);
    int blocks = ceil_div(waves, 8);
    gemm_wmma_frag<<<blocks, T, 0, stream>>>((const v4u*)h1h, (const v4u*)h1l,
                                             (const v4u*)wl2h, (const v4u*)wl2l,
                                             xl2, KN, KC2, KHC);
    gemm_wmma_frag<<<blocks, T, 0, stream>>>((const v4u*)h1h, (const v4u*)h1l,
                                             (const v4u*)wr2h, (const v4u*)wr2l,
                                             xr2, KN, KC2, KHC);
  }

  // --- layer 2 edge phase ---
  fillk<<<ceil_div(KN, T), T, 0, stream>>>(smax2, -INFINITY, KN);
  fillk<<<ceil_div(KN, T), T, 0, stream>>>(den2, 0.f, KN);
  fillk<<<ceil_div((long long)KN * KC2, T), T, 0, stream>>>(out2, 0.f, KN * KC2);
  {
    int eb = ceil_div((long long)KETOT * 32, T);
    edge_score<KC2, 1><<<eb, T, 0, stream>>>(xl2, xr2, att2, ei, sc2, smax2);
    edge_exp<1><<<ceil_div(KETOT, T), T, 0, stream>>>(sc2, smax2, den2, ei);
    edge_aggr<KC2, 1><<<eb, T, 0, stream>>>(sc2, den2, xl2, ei, out2);
  }

  // --- pool + output head ---
  fillk<<<1, T, 0, stream>>>(pooled, 0.f, KG * KC2);
  fillk<<<1, 32, 0, stream>>>(cnt, 0.f, KG);
  pool_accum<<<ceil_div((long long)KN * 64, T), T, 0, stream>>>(out2, b2, batc, pooled, cnt);
  final_head<<<1, 256, 0, stream>>>(pooled, cnt, Wo, bo, out);
}
